// AtenMmQuint8_54752243089863
// MI455X (gfx1250) — 8-bit matrix operand order repaired and checked by reading, compiled, not run on hardware
//
#include <hip/hip_runtime.h>

#define M_DIM 4096
#define N_DIM 4096
#define K_DIM 4096

#define A_STRIDE 80
#define B_STRIDE 80

typedef __attribute__((ext_vector_type(8))) int v8i;
typedef __attribute__((ext_vector_type(4))) int v4i;

#if __has_builtin(__builtin_amdgcn_global_load_async_to_lds_b128)
#define HAVE_ASYNC_LDS 1
#endif

__global__ void pack_x_kernel(const int* __restrict__ x, unsigned char* __restrict__ xp) {
    int idx = blockIdx.x * blockDim.x + threadIdx.x;
    int4 v = ((const int4*)x)[idx];
    unsigned int b = (unsigned)(v.x & 255) | ((unsigned)(v.y & 255) << 8) |
                     ((unsigned)(v.z & 255) << 16) | ((unsigned)(v.w & 255) << 24);
    ((unsigned int*)xp)[idx] = b;
}

__global__ void transpose_pack_y_kernel(const int* __restrict__ y, unsigned char* __restrict__ yt) {
    __shared__ unsigned char tile[64 * 65];
    const int t = threadIdx.x;
    const int tileN = blockIdx.x * 64;
    const int tileK = blockIdx.y * 64;
#pragma unroll
    for (int j = 0; j < 16; ++j) {
        int idx = t + j * 256;
        int r = idx >> 6, c = idx & 63;
        int val = y[(size_t)(tileK + r) * N_DIM + tileN + c];
        tile[r * 65 + c] = (unsigned char)val;
    }
    __syncthreads();
#pragma unroll
    for (int j = 0; j < 4; ++j) {
        int d = t + j * 256;
        int byteIdx = d << 2;
        int nl = byteIdx >> 6, kl = byteIdx & 63;
        unsigned int w = (unsigned)tile[(kl + 0) * 65 + nl] |
                         ((unsigned)tile[(kl + 1) * 65 + nl] << 8) |
                         ((unsigned)tile[(kl + 2) * 65 + nl] << 16) |
                         ((unsigned)tile[(kl + 3) * 65 + nl] << 24);
        *(unsigned int*)&yt[(size_t)(tileN + nl) * K_DIM + tileK + kl] = w;
    }
}

__global__ void rowsum_x_kernel(const int* __restrict__ x, int* __restrict__ rs) {
    __shared__ int sm[256];
    const int row = blockIdx.x;
    const int t = threadIdx.x;
    int s = 0;
#pragma unroll
    for (int j = 0; j < 16; ++j) s += x[(size_t)row * K_DIM + t + j * 256];
    sm[t] = s;
    __syncthreads();
    for (int o = 128; o > 0; o >>= 1) {
        if (t < o) sm[t] += sm[t + o];
        __syncthreads();
    }
    if (t == 0) rs[row] = sm[0];
}

__global__ void colsum_y_kernel(const int* __restrict__ y, int* __restrict__ cs) {
    const int n = blockIdx.x * 256 + threadIdx.x;
    const int k0 = blockIdx.y * 128;
    int s = 0;
#pragma unroll 4
    for (int k = 0; k < 128; ++k) s += y[(size_t)(k0 + k) * N_DIM + n];
    atomicAdd(&cs[n], s);
}

__device__ __forceinline__ void copy16(const unsigned char* g, unsigned char* l) {
#ifdef HAVE_ASYNC_LDS
    typedef int vec4 __attribute__((ext_vector_type(4)));
    typedef __attribute__((address_space(1))) vec4* global_v4;
    typedef __attribute__((address_space(3))) vec4* lds_v4;
    __builtin_amdgcn_global_load_async_to_lds_b128((global_v4)g, (lds_v4)l, 0, 0);
#else
    *(int4*)l = *(const int4*)g;
#endif
}

template <int N>
__device__ __forceinline__ void wait_async() {
#ifdef HAVE_ASYNC_LDS
#if __has_builtin(__builtin_amdgcn_s_wait_asynccnt)
    __builtin_amdgcn_s_wait_asynccnt(N);
#else
    asm volatile("s_wait_asynccnt %0" ::"i"(N) : "memory");
#endif
#endif
}

__global__ void __launch_bounds__(256)
gemm_u8_kernel(const unsigned char* __restrict__ xp,
               const unsigned char* __restrict__ ytp,
               const int* __restrict__ rowSum,
               const int* __restrict__ colSum,
               float* __restrict__ out) {
    __shared__ __align__(16) unsigned char As[2][128 * A_STRIDE];
    __shared__ __align__(16) unsigned char Bs[2][256 * B_STRIDE];

    const int t = threadIdx.x;
    const int lane = t & 31;
    const int wave = t >> 5;
    const int waveM = wave >> 1;
    const int waveN = wave & 1;
    const int l15 = lane & 15;
    const int lhi = lane >> 4;
    const int mBase = blockIdx.y * 128;
    const int nBase = blockIdx.x * 256;

    v8i acc[2][8] = {};

    auto load_tiles = [&](int buf, int kt) {
#pragma unroll
        for (int it = 0; it < 2; ++it) {
            int c = t + it * 256;
            int r = c >> 2, off = (c & 3) << 4;
            copy16(&xp[(size_t)(mBase + r) * K_DIM + kt + off], &As[buf][r * A_STRIDE + off]);
        }
#pragma unroll
        for (int it = 0; it < 4; ++it) {
            int c = t + it * 256;
            int r = c >> 2, off = (c & 3) << 4;
            copy16(&ytp[(size_t)(nBase + r) * K_DIM + kt + off], &Bs[buf][r * B_STRIDE + off]);
        }
    };

    auto loadB = [&](int buf, int nf) -> v8i {
        int col = waveN * 128 + nf * 16 + l15;
        const v4i* q = (const v4i*)&Bs[buf][col * B_STRIDE + lhi * 16];
        v4i b0 = q[0];
        v4i b1 = q[2];
        v8i b;
        b[0] = b0[0]; b[1] = b0[1]; b[2] = b0[2]; b[3] = b0[3];
        b[4] = b1[0]; b[5] = b1[1]; b[6] = b1[2]; b[7] = b1[3];
        return b;
    };

    load_tiles(0, 0);

    for (int kt = 0; kt < K_DIM; kt += 64) {
        const int buf = (kt >> 6) & 1;
        __syncthreads();
        const bool more = (kt + 64) < K_DIM;
        if (more) {
            load_tiles(buf ^ 1, kt + 64);
            wait_async<6>();
        } else {
            wait_async<0>();
        }
        __syncthreads();

        v8i afrag[2];
#pragma unroll
        for (int mf = 0; mf < 2; ++mf) {
            int row = waveM * 32 + mf * 16 + l15;
            const unsigned int* p = (const unsigned int*)&As[buf][row * A_STRIDE + lhi * 16];
            v8i a;
            a[0] = (int)p[0];  a[1] = (int)p[1];
            a[2] = (int)p[2];  a[3] = (int)p[3];
            a[4] = (int)p[8];  a[5] = (int)p[9];
            a[6] = (int)p[10]; a[7] = (int)p[11];
            afrag[mf] = a;
        }

        v8i bcur = loadB(buf, 0);
#pragma unroll
        for (int nf = 0; nf < 8; ++nf) {
            v8i bnext;
            if (nf < 7) bnext = loadB(buf, nf + 1);
            acc[0][nf] = __builtin_amdgcn_wmma_i32_16x16x64_iu8(
                false, afrag[0], false, bcur, acc[0][nf], false, false);
            acc[1][nf] = __builtin_amdgcn_wmma_i32_16x16x64_iu8(
                false, afrag[1], false, bcur, acc[1][nf], false, false);
            if (nf < 7) bcur = bnext;
        }
    }

    const float SCALE = 0.199f * 0.0215f;
    const int KZZ = K_DIM * 65 * 160;
#pragma unroll
    for (int mf = 0; mf < 2; ++mf) {
        int mFragBase = mBase + waveM * 32 + mf * 16 + lhi * 8;
        int rs[8];
#pragma unroll
        for (int v = 0; v < 8; ++v) rs[v] = rowSum[mFragBase + v];
#pragma unroll
        for (int nf = 0; nf < 8; ++nf) {
            int n = nBase + waveN * 128 + nf * 16 + l15;
            int csum = colSum[n];
#pragma unroll
            for (int v = 0; v < 8; ++v) {
                int m = mFragBase + v;
                int corrected = acc[mf][nf][v] - 65 * csum - 160 * rs[v] + KZZ;
                out[(size_t)m * N_DIM + n] = SCALE * (float)corrected;
            }
        }
    }
}

extern "C" void kernel_launch(void* const* d_in, const int* in_sizes, int n_in,
                              void* d_out, int out_size, void* d_ws, size_t ws_size,
                              hipStream_t stream) {
    const int* x = (const int*)d_in[0];
    const int* y = (const int*)d_in[1];
    float* out = (float*)d_out;

    unsigned char* xp  = (unsigned char*)d_ws;
    unsigned char* ytp = xp + (size_t)M_DIM * K_DIM;
    int* rowSum = (int*)(ytp + (size_t)N_DIM * K_DIM);
    int* colSum = rowSum + M_DIM;

    (void)hipMemsetAsync(colSum, 0, N_DIM * sizeof(int), stream);

    pack_x_kernel<<<(M_DIM * K_DIM / 4) / 256, 256, 0, stream>>>(x, xp);
    transpose_pack_y_kernel<<<dim3(N_DIM / 64, K_DIM / 64), 256, 0, stream>>>(y, ytp);
    rowsum_x_kernel<<<M_DIM, 256, 0, stream>>>(x, rowSum);
    colsum_y_kernel<<<dim3(N_DIM / 256, K_DIM / 128), 256, 0, stream>>>(y, colSum);

    gemm_u8_kernel<<<dim3(N_DIM / 256, M_DIM / 128), 256, 0, stream>>>(
        xp, ytp, rowSum, colSum, out);
}
